// ComplexNetAttention_86208583565962
// MI455X (gfx1250) — compile-verified, matrix operand order checked by reading, not run on hardware
//
#include <hip/hip_runtime.h>
#include <hip/hip_bf16.h>

// ---------------------------------------------------------------------------
// ComplexNet attention for MI455X (gfx1250, wave32, WMMA).
// Heavy math: v_wmma_f32_16x16x32_f16.
// Data movement: global_load_async_to_lds_b128 (ASYNCcnt) for activations,
// tensor_load_to_lds (TDM, TENSORcnt) for weight panels.
// ---------------------------------------------------------------------------

#define T_DIM 2048
#define H_DIM 2048
#define NHEAD 16
#define HDIM  128   // head dim (per real/imag component)

typedef __attribute__((ext_vector_type(16))) _Float16 v16h;
typedef __attribute__((ext_vector_type(8)))  float    v8f;
typedef __attribute__((ext_vector_type(4)))  int      i32x4;
typedef __attribute__((ext_vector_type(8)))  int      i32x8;

union FragH { v16h v; float4 q[2]; unsigned int u[8]; };

// ---------------------------------------------------------------------------
// CDNA5 async / tensor data movers (inline asm: portable across toolchains).
// ---------------------------------------------------------------------------
__device__ __forceinline__ unsigned lds_off_u32(const void* p) {
    // Flat LDS addressing uses addr[31:0] as the LDS byte address.
    return (unsigned)(unsigned long long)p;
}

// Per-lane 16B global -> LDS async copy (GLOBAL_LOAD_ASYNC_TO_LDS_B128).
__device__ __forceinline__ void async_copy_b128(void* lds_dst, const void* gsrc) {
    asm volatile("global_load_async_to_lds_b128 %0, %1, off"
                 :: "v"(lds_off_u32(lds_dst)), "v"(gsrc)
                 : "memory");
}
__device__ __forceinline__ void wait_async_all() {
    asm volatile("s_wait_asynccnt 0x0" ::: "memory");
}

// TDM: 2D tile of 2-byte elements, global -> LDS, with LDS row padding.
// tile_d0 must make a multiple of 4 bytes; pad fields fixed for a
// 32-half tile row padded to a 40-half LDS pitch (16 DWORDs + 4 DWORDs pad).
__device__ __forceinline__ void tdm_load_2d_f16(unsigned lds_byte_off, const void* gaddr,
                                                unsigned tile_d0, unsigned tile_d1,
                                                unsigned tensor_d0, unsigned tensor_d1,
                                                unsigned stride_d0) {
    const unsigned long long ga = (unsigned long long)gaddr;
    i32x4 g0;
    g0[0] = 1;                                        // count=1, user descriptor
    g0[1] = (int)lds_byte_off;                        // lds_addr
    g0[2] = (int)(unsigned)(ga & 0xffffffffull);      // global_addr[31:0]
    g0[3] = (int)((unsigned)(ga >> 32) | (2u << 30)); // global_addr[56:32] | type=2
    i32x8 g1;
    g1[0] = (1 << 16)   // data_size = 2 bytes
          | (1 << 20)   // pad_enable
          | (3 << 22)   // pad_interval: 16 DWORDs between pads
          | (3 << 25);  // pad_amount: 4 DWORDs
    g1[1] = (int)(tensor_d0 << 16);                          // tensor_dim0 -> bits 79:48
    g1[2] = (int)((tensor_d0 >> 16) | (tensor_d1 << 16));    // tensor_dim1 -> bits 111:80
    g1[3] = (int)((tensor_d1 >> 16) | (tile_d0 << 16));      // tile_dim0 -> bits 127:112
    g1[4] = (int)(tile_d1 & 0xffff);                         // tile_dim1 (tile_dim2=0)
    g1[5] = (int)stride_d0;                                  // tensor_dim0_stride[31:0]
    g1[6] = 0;                                               // stride hi / dim1_stride lo
    g1[7] = 0;
    asm volatile("tensor_load_to_lds %0, %1" :: "s"(g0), "s"(g1) : "memory");
}
__device__ __forceinline__ void wait_tensor_all() {
    asm volatile("s_wait_tensorcnt 0x0" ::: "memory");
}

// ---------------------------------------------------------------------------
// WMMA fragment helpers.
// Per CDNA5 ISA layout (16-bit A 16x32): lane l (l<16) row=l holds K 0..7 in
// v[0..3] (pairs) and K 16..23 in v[4..7]; lanes 16..31 row=l-16 hold K 8..15
// and 24..31. B (stored as Bt rows = columns of B) uses the identical pattern.
// ---------------------------------------------------------------------------
__device__ __forceinline__ v16h load_frag(const _Float16* base, int ld, int lane) {
    const int r  = lane & 15;
    const int ko = (lane >> 4) << 3;   // 0 or 8
    FragH f;
    f.q[0] = *reinterpret_cast<const float4*>(base + (size_t)r * ld + ko);
    f.q[1] = *reinterpret_cast<const float4*>(base + (size_t)r * ld + ko + 16);
    return f.v;
}

__device__ __forceinline__ v16h neg_frag(v16h x) {
    FragH f; f.v = x;
#pragma unroll
    for (int i = 0; i < 8; ++i) f.u[i] ^= 0x80008000u;  // flip packed f16 signs
    return f.v;
}

__device__ __forceinline__ v8f wmma_f16(v16h a, v16h b, v8f c) {
    return __builtin_amdgcn_wmma_f32_16x16x32_f16(false, a, false, b, (short)0, c,
                                                  false, false);
}

// ---------------------------------------------------------------------------
// Per-row symmetric int8 fake-quant (matches ActivationQuantSTE) -> f16 out.
// ---------------------------------------------------------------------------
__global__ __launch_bounds__(256)
void cnq_quantize_rows(const float* __restrict__ x, _Float16* __restrict__ y) {
    __shared__ float red[8];
    const int row = blockIdx.x;
    const float* xr = x + (size_t)row * H_DIM;
    float am = 0.0f;
    for (int c = threadIdx.x; c < H_DIM; c += 256)
        am = fmaxf(am, fabsf(xr[c]));
#pragma unroll
    for (int off = 1; off < 32; off <<= 1)
        am = fmaxf(am, __shfl_xor(am, off, 32));
    if ((threadIdx.x & 31) == 0) red[threadIdx.x >> 5] = am;
    __syncthreads();
    float m = red[0];
#pragma unroll
    for (int i = 1; i < 8; ++i) m = fmaxf(m, red[i]);
    const float scl  = 127.0f / fmaxf(m, 1e-5f);
    const float iscl = 1.0f / scl;
    for (int c = threadIdx.x; c < H_DIM; c += 256) {
        float q = rintf(xr[c] * scl);                 // RNE, like jnp.round
        q = fminf(fmaxf(q, -128.0f), 127.0f);
        y[(size_t)row * H_DIM + c] = (_Float16)(q * iscl);
    }
}

// ---------------------------------------------------------------------------
// fp32 -> f16 elementwise (weights).
// ---------------------------------------------------------------------------
__global__ __launch_bounds__(256)
void cnq_f32_to_f16(const float* __restrict__ x, _Float16* __restrict__ y, int n) {
    int i = blockIdx.x * 256 + threadIdx.x;
    if (i < n) y[i] = (_Float16)x[i];
}

// ---------------------------------------------------------------------------
// Complex GEMM:  Yr = Ar@Wr^T + Ai@Wi^T ;  Yi = Ar@Wi^T - Ai@Wr^T
// Block tile 128x128, 256 threads = 8 waves, each wave 32x64.
// A panels: async-to-LDS; W panels: TDM with padded LDS pitch.
// ---------------------------------------------------------------------------
__global__ __launch_bounds__(256)
void cnq_complex_gemm(const _Float16* __restrict__ Ar, const _Float16* __restrict__ Ai,
                      const _Float16* __restrict__ Wr, const _Float16* __restrict__ Wi,
                      float* __restrict__ Yr, float* __restrict__ Yi) {
    __shared__ _Float16 sAr[128 * 40];
    __shared__ _Float16 sAi[128 * 40];
    __shared__ _Float16 sBr[128 * 40];
    __shared__ _Float16 sBi[128 * 40];

    const int tid  = threadIdx.x;
    const int lane = tid & 31;
    const int w    = tid >> 5;
    const int m0   = blockIdx.y * 128;
    const int n0   = blockIdx.x * 128;
    const int wm   = (w & 3) * 32;   // wave row offset inside block tile
    const int wn   = (w >> 2) * 64;  // wave col offset inside block tile

    v8f accR[2][4], accI[2][4];
    const v8f vzero = {};
#pragma unroll
    for (int a = 0; a < 2; ++a)
#pragma unroll
        for (int b = 0; b < 4; ++b) { accR[a][b] = vzero; accI[a][b] = vzero; }

    for (int kk = 0; kk < H_DIM; kk += 32) {
        __syncthreads();

        // A panels: 128 rows x 32 halves = 512 16B chunks across both tiles;
        // one async b128 per lane per chunk.
#pragma unroll
        for (int c = tid; c < 512; c += 256) {
            const int r  = c >> 2;
            const int kc = (c & 3) * 8;
            const size_t ga = (size_t)(m0 + r) * H_DIM + kk + kc;
            async_copy_b128(&sAr[r * 40 + kc], &Ar[ga]);
            async_copy_b128(&sAi[r * 40 + kc], &Ai[ga]);
        }

        // W panels via Tensor Data Mover: one wave drives both descriptors.
        if (w == 0) {
            tdm_load_2d_f16(lds_off_u32(sBr), &Wr[(size_t)n0 * H_DIM + kk],
                            /*tile*/32, 128, /*tensor*/H_DIM, H_DIM, /*stride*/H_DIM);
            tdm_load_2d_f16(lds_off_u32(sBi), &Wi[(size_t)n0 * H_DIM + kk],
                            32, 128, H_DIM, H_DIM, H_DIM);
            wait_tensor_all();
        }
        wait_async_all();
        __syncthreads();

        if (kk + 32 < H_DIM) {  // warm L2 for the next K panel (global_prefetch_b8)
            __builtin_prefetch(&Ar[(size_t)(m0 + (tid >> 1)) * H_DIM + kk + 32], 0, 1);
            __builtin_prefetch(&Wr[(size_t)(n0 + (tid >> 1)) * H_DIM + kk + 32], 0, 1);
        }

        const v16h a_r0 = load_frag(sAr + (wm +  0) * 40, 40, lane);
        const v16h a_r1 = load_frag(sAr + (wm + 16) * 40, 40, lane);
        const v16h a_i0 = load_frag(sAi + (wm +  0) * 40, 40, lane);
        const v16h a_i1 = load_frag(sAi + (wm + 16) * 40, 40, lane);

#pragma unroll
        for (int ns = 0; ns < 4; ++ns) {
            const v16h br  = load_frag(sBr + (wn + ns * 16) * 40, 40, lane);
            const v16h bi  = load_frag(sBi + (wn + ns * 16) * 40, 40, lane);
            const v16h brn = neg_frag(br);   // f16 WMMA has no A/B neg bits; flip signs
            accR[0][ns] = wmma_f16(a_r0, br,  accR[0][ns]);
            accR[0][ns] = wmma_f16(a_i0, bi,  accR[0][ns]);
            accI[0][ns] = wmma_f16(a_r0, bi,  accI[0][ns]);
            accI[0][ns] = wmma_f16(a_i0, brn, accI[0][ns]);
            accR[1][ns] = wmma_f16(a_r1, br,  accR[1][ns]);
            accR[1][ns] = wmma_f16(a_i1, bi,  accR[1][ns]);
            accI[1][ns] = wmma_f16(a_r1, bi,  accI[1][ns]);
            accI[1][ns] = wmma_f16(a_i1, brn, accI[1][ns]);
        }
    }

    // C layout: VGPR r -> rows r (lanes 0..15) / r+8 (lanes 16..31), col = lane&15
    const int colL = lane & 15;
    const int hi   = lane >> 4;
#pragma unroll
    for (int ms = 0; ms < 2; ++ms)
#pragma unroll
        for (int ns = 0; ns < 4; ++ns)
#pragma unroll
            for (int r = 0; r < 8; ++r) {
                const int row = m0 + wm + ms * 16 + r + hi * 8;
                const int col = n0 + wn + ns * 16 + colL;
                Yr[(size_t)row * H_DIM + col] = accR[ms][ns][r];
                Yi[(size_t)row * H_DIM + col] = accI[ms][ns][r];
            }
}

// ---------------------------------------------------------------------------
// RoPE (complex rotation, bf16-roundtripped cos/sin) + f16 conversion of V.
// ---------------------------------------------------------------------------
__global__ __launch_bounds__(256)
void cnq_rope_convert(const float* __restrict__ qr, const float* __restrict__ qi,
                      const float* __restrict__ kr, const float* __restrict__ ki,
                      const float* __restrict__ vr, const float* __restrict__ vi,
                      const int* __restrict__ pos,
                      _Float16* __restrict__ qr16, _Float16* __restrict__ qi16,
                      _Float16* __restrict__ kr16, _Float16* __restrict__ ki16,
                      _Float16* __restrict__ vr16, _Float16* __restrict__ vi16) {
    const size_t idx = (size_t)blockIdx.x * 256 + threadIdx.x;  // over T*H
    const int t = (int)(idx >> 11);
    const int d = (int)(idx & (HDIM - 1));
    const float p    = (float)pos[t];
    const float invf = powf(10000.0f, -(float)d * (1.0f / (float)HDIM));
    const float fr   = p * invf;
    const float c = __bfloat162float(__float2bfloat16(cosf(fr)));
    const float s = __bfloat162float(__float2bfloat16(sinf(fr)));

    const float xqr = qr[idx], xqi = qi[idx];
    qr16[idx] = (_Float16)(xqr * c - xqi * s);
    qi16[idx] = (_Float16)(xqi * c + xqr * s);
    const float xkr = kr[idx], xki = ki[idx];
    kr16[idx] = (_Float16)(xkr * c - xki * s);
    ki16[idx] = (_Float16)(xki * c + xkr * s);
    vr16[idx] = (_Float16)vr[idx];
    vi16[idx] = (_Float16)vi[idx];
}

// ---------------------------------------------------------------------------
// Flash attention, one workgroup per (64-query block, head).
// 4 waves x 16 query rows. Scores reduce over K-dim 256 = [qr|qi]·[kr|ki].
// scale = 1/sqrt(2*D) = 1/16. Online softmax; P staged via LDS; PV via WMMA.
// Q/K panels arrive by async-to-LDS copies.
// ---------------------------------------------------------------------------
__global__ __launch_bounds__(128)
void cnq_flash_attn(const _Float16* __restrict__ qr, const _Float16* __restrict__ qi,
                    const _Float16* __restrict__ kr, const _Float16* __restrict__ ki,
                    const _Float16* __restrict__ vr, const _Float16* __restrict__ vi,
                    float* __restrict__ outR, float* __restrict__ outI) {
    __shared__ _Float16 sQ[64 * 264];    // [row][0:128 qr | 128:256 qi]
    __shared__ _Float16 sK[64 * 264];
    __shared__ _Float16 sVr[128 * 72];   // transposed: [d][key]
    __shared__ _Float16 sVi[128 * 72];
    __shared__ _Float16 sP[64 * 72];     // per-wave 16x64 strips

    const int tid  = threadIdx.x;
    const int lane = tid & 31;
    const int w    = tid >> 5;
    const int qb   = blockIdx.x;
    const int h    = blockIdx.y;
    const int qrow0 = qb * 64;
    const int colL = lane & 15;
    const int hi   = lane >> 4;

    // Q block: 64 rows x 32 8-half chunks (16 real + 16 imag), async to LDS.
    for (int c = tid; c < 2048; c += 128) {
        const int r  = c >> 5;
        const int cc = c & 31;
        const _Float16* src = (cc < 16)
            ? &qr[(size_t)(qrow0 + r) * H_DIM + h * HDIM + cc * 8]
            : &qi[(size_t)(qrow0 + r) * H_DIM + h * HDIM + (cc - 16) * 8];
        async_copy_b128(&sQ[r * 264 + cc * 8], src);
    }

    v8f oR[8], oI[8];
    float mrun[8], lrun[8];
    const v8f vzero = {};
#pragma unroll
    for (int i = 0; i < 8; ++i) { oR[i] = vzero; oI[i] = vzero; mrun[i] = -1e30f; lrun[i] = 0.0f; }
    const float scale = 0.0625f;  // 1/sqrt(256)

    for (int kb = 0; kb <= qb; ++kb) {
        __syncthreads();  // previous PV done before overwriting K/V
        for (int c = tid; c < 2048; c += 128) {
            const int r  = c >> 5;
            const int cc = c & 31;
            const _Float16* src = (cc < 16)
                ? &kr[(size_t)(kb * 64 + r) * H_DIM + h * HDIM + cc * 8]
                : &ki[(size_t)(kb * 64 + r) * H_DIM + h * HDIM + (cc - 16) * 8];
            async_copy_b128(&sK[r * 264 + cc * 8], src);
        }
        for (int e = tid; e < 8192; e += 128) {   // V transposed into LDS
            const int d   = e >> 6;
            const int key = e & 63;
            sVr[d * 72 + key] = vr[(size_t)(kb * 64 + key) * H_DIM + h * HDIM + d];
            sVi[d * 72 + key] = vi[(size_t)(kb * 64 + key) * H_DIM + h * HDIM + d];
        }
        wait_async_all();   // Q (first iter) + K panels committed to LDS
        __syncthreads();

        // S = Q_concat x K_concat^T  (16 rows x 64 keys per wave)
        v8f S[4];
#pragma unroll
        for (int ns = 0; ns < 4; ++ns) {
            v8f s = vzero;
#pragma unroll
            for (int kc = 0; kc < 8; ++kc) {
                const v16h a = load_frag(sQ + (w * 16) * 264 + kc * 32, 264, lane);
                const v16h b = load_frag(sK + (ns * 16) * 264 + kc * 32, 264, lane);
                s = wmma_f16(a, b, s);
            }
            S[ns] = s;
        }

        // scale + causal mask
#pragma unroll
        for (int ns = 0; ns < 4; ++ns)
#pragma unroll
            for (int r = 0; r < 8; ++r) {
                float v = S[ns][r] * scale;
                const int key = kb * 64 + ns * 16 + colL;
                const int qpos = qrow0 + w * 16 + r + hi * 8;
                if (key > qpos) v = -1e30f;
                S[ns][r] = v;
            }

        // online softmax: row stats live per 16-lane group (matches C layout)
        float alpha[8];
#pragma unroll
        for (int r = 0; r < 8; ++r) {
            float t = fmaxf(fmaxf(S[0][r], S[1][r]), fmaxf(S[2][r], S[3][r]));
#pragma unroll
            for (int off = 1; off < 16; off <<= 1)
                t = fmaxf(t, __shfl_xor(t, off, 16));
            const float mn = fmaxf(mrun[r], t);
            alpha[r] = __expf(mrun[r] - mn);
            mrun[r]  = mn;
        }
        float rsum[8];
#pragma unroll
        for (int r = 0; r < 8; ++r) rsum[r] = 0.0f;
#pragma unroll
        for (int ns = 0; ns < 4; ++ns)
#pragma unroll
            for (int r = 0; r < 8; ++r) {
                const float p = __expf(S[ns][r] - mrun[r]);
                S[ns][r] = p;
                rsum[r] += p;
            }
#pragma unroll
        for (int r = 0; r < 8; ++r) {
            float t = rsum[r];
#pragma unroll
            for (int off = 1; off < 16; off <<= 1)
                t += __shfl_xor(t, off, 16);
            lrun[r] = lrun[r] * alpha[r] + t;
        }
#pragma unroll
        for (int nv = 0; nv < 8; ++nv)
#pragma unroll
            for (int r = 0; r < 8; ++r) {
                oR[nv][r] = oR[nv][r] * alpha[r];
                oI[nv][r] = oI[nv][r] * alpha[r];
            }

        // stage P (f16) into this wave's LDS strip
#pragma unroll
        for (int ns = 0; ns < 4; ++ns)
#pragma unroll
            for (int r = 0; r < 8; ++r)
                sP[(w * 16 + r + hi * 8) * 72 + ns * 16 + colL] = (_Float16)S[ns][r];
        __syncthreads();

        // O += P x V  (K-dim = 64 keys -> two wmma steps)
#pragma unroll
        for (int nv = 0; nv < 8; ++nv) {
#pragma unroll
            for (int kc = 0; kc < 2; ++kc) {
                const v16h a  = load_frag(sP + (w * 16) * 72 + kc * 32, 72, lane);
                const v16h br = load_frag(sVr + (nv * 16) * 72 + kc * 32, 72, lane);
                const v16h bi = load_frag(sVi + (nv * 16) * 72 + kc * 32, 72, lane);
                oR[nv] = wmma_f16(a, br, oR[nv]);
                oI[nv] = wmma_f16(a, bi, oI[nv]);
            }
        }
    }

#pragma unroll
    for (int nv = 0; nv < 8; ++nv)
#pragma unroll
        for (int r = 0; r < 8; ++r) {
            const float inv = 1.0f / lrun[r];
            const int row = qrow0 + w * 16 + r + hi * 8;
            const int col = h * HDIM + nv * 16 + colL;
            outR[(size_t)row * H_DIM + col] = oR[nv][r] * inv;
            outI[(size_t)row * H_DIM + col] = oI[nv][r] * inv;
        }
}

// ---------------------------------------------------------------------------
// Host orchestration.
// ---------------------------------------------------------------------------
extern "C" void kernel_launch(void* const* d_in, const int* in_sizes, int n_in,
                              void* d_out, int out_size, void* d_ws, size_t ws_size,
                              hipStream_t stream) {
    (void)in_sizes; (void)n_in; (void)out_size; (void)ws_size;
    const size_t TH = (size_t)T_DIM * H_DIM;

    const float* hid_r = (const float*)d_in[0];
    const float* hid_i = (const float*)d_in[1];
    const int*   pos   = (const int*)d_in[2];
    const float* wsrc[8] = {
        (const float*)d_in[3], (const float*)d_in[4],   // Wq_r, Wq_i
        (const float*)d_in[5], (const float*)d_in[6],   // Wk_r, Wk_i
        (const float*)d_in[7], (const float*)d_in[8],   // Wv_r, Wv_i
        (const float*)d_in[9], (const float*)d_in[10],  // Wo_r, Wo_i
    };

    char* ws = (char*)d_ws;
    size_t off = 0;
    auto allocH = [&](size_t n) { void* p = ws + off; off += n * sizeof(_Float16); return (_Float16*)p; };
    auto allocF = [&](size_t n) { void* p = ws + off; off += n * sizeof(float);    return (float*)p; };

    _Float16* hAr = allocH(TH);
    _Float16* hAi = allocH(TH);
    _Float16* W16 = allocH(8 * TH);
    float* fQr = allocF(TH); float* fQi = allocF(TH);
    float* fKr = allocF(TH); float* fKi = allocF(TH);
    float* fVr = allocF(TH); float* fVi = allocF(TH);
    _Float16* qR16 = allocH(TH); _Float16* qI16 = allocH(TH);
    _Float16* kR16 = allocH(TH); _Float16* kI16 = allocH(TH);
    _Float16* vR16 = allocH(TH); _Float16* vI16 = allocH(TH);
    float* aR = allocF(TH); float* aI = allocF(TH);
    _Float16* oR16 = allocH(TH); _Float16* oI16 = allocH(TH);

    // 1. activation fake-quant -> f16 (shared by q/k/v projections)
    cnq_quantize_rows<<<T_DIM, 256, 0, stream>>>(hid_r, hAr);
    cnq_quantize_rows<<<T_DIM, 256, 0, stream>>>(hid_i, hAi);

    // 2. weights fp32 -> f16
    for (int i = 0; i < 8; ++i)
        cnq_f32_to_f16<<<(int)(TH / 256), 256, 0, stream>>>(wsrc[i], W16 + (size_t)i * TH, (int)TH);

    // 3. complex QKV projections (WMMA + async/TDM staging)
    dim3 ggrid(H_DIM / 128, T_DIM / 128);
    cnq_complex_gemm<<<ggrid, 256, 0, stream>>>(hAr, hAi, W16 + 0 * TH, W16 + 1 * TH, fQr, fQi);
    cnq_complex_gemm<<<ggrid, 256, 0, stream>>>(hAr, hAi, W16 + 2 * TH, W16 + 3 * TH, fKr, fKi);
    cnq_complex_gemm<<<ggrid, 256, 0, stream>>>(hAr, hAi, W16 + 4 * TH, W16 + 5 * TH, fVr, fVi);

    // 4. RoPE + f16 conversion
    cnq_rope_convert<<<(int)(TH / 256), 256, 0, stream>>>(
        fQr, fQi, fKr, fKi, fVr, fVi, pos,
        qR16, qI16, kR16, kI16, vR16, vI16);

    // 5. causal flash attention (WMMA)
    dim3 fgrid(T_DIM / 64, NHEAD);
    cnq_flash_attn<<<fgrid, 128, 0, stream>>>(qR16, qI16, kR16, kI16, vR16, vI16, aR, aI);

    // 6. output projection: quant -> complex GEMM into d_out = [yr | yi] (f32)
    cnq_quantize_rows<<<T_DIM, 256, 0, stream>>>(aR, oR16);
    cnq_quantize_rows<<<T_DIM, 256, 0, stream>>>(aI, oI16);
    float* yout = (float*)d_out;
    cnq_complex_gemm<<<ggrid, 256, 0, stream>>>(oR16, oI16, W16 + 6 * TH, W16 + 7 * TH,
                                                yout, yout + TH);
}
